// EncoderLayer_21354577396127
// MI455X (gfx1250) — compile-verified, matrix operand order checked by reading, not run on hardware
//
#include <hip/hip_runtime.h>
#include <hip/hip_bf16.h>

#define N_NODES 10000
#define T_STEPS 12
#define D_FEAT  64
#define H_HEADS 4
#define DH      16
#define E_EDGES 100000
#define DFF     128
#define NT      (N_NODES * T_STEPS)      /* 120000 */
#define NTD     ((size_t)NT * D_FEAT)    /* 7.68M  */
#define NTH     (NT * H_HEADS)           /* 480000 */
#define ET      (E_EDGES * T_STEPS)      /* 1.2M   */
#define STRIPS  (NT / 16)                /* 7500   */
#define TD      (T_STEPS * D_FEAT)       /* 768    */
#define BN_EPS  1e-5f
#define SLOPE   0.2f

typedef __attribute__((ext_vector_type(16))) _Float16 v16h;
typedef __attribute__((ext_vector_type(8)))  _Float16 v8h;
typedef __attribute__((ext_vector_type(8)))  float    v8f;

// ---------------- order-preserving float <-> uint for atomic max ----------
__device__ __forceinline__ unsigned f2ord(float f) {
    unsigned b = __float_as_uint(f);
    return (b & 0x80000000u) ? ~b : (b | 0x80000000u);
}
__device__ __forceinline__ float ord2f(unsigned u) {
    unsigned b = (u & 0x80000000u) ? (u & 0x7fffffffu) : ~u;
    return __uint_as_float(b);
}
#define ORD_NEG_INF 0x007FFFFFu   /* f2ord(-inf) */

// ---------------- WMMA fragment loaders (wave32, 16-bit operands) ---------
// A 16x32 (MxK): lane l -> row = l&15, hi = l>>4; element i (0..15):
//   i<8 : K = hi*8 + i            (contiguous 8 halves -> one b128 load)
//   i>=8: K = 16 + hi*8 + (i-8)   (second contiguous b128 load)
__device__ __forceinline__ v16h load_a_frag(const _Float16* A, int ld, int row0, int k0) {
    int lane = threadIdx.x & 31;
    int m = lane & 15, hi = lane >> 4;
    const _Float16* p0 = A + (size_t)(row0 + m) * ld + k0 + hi * 8;
    v8h a0 = *(const v8h*)(p0);
    v8h a1 = *(const v8h*)(p0 + 16);
    v16h r;
#pragma unroll
    for (int i = 0; i < 8; ++i) { r[i] = a0[i]; r[i + 8] = a1[i]; }
    return r;
}

// B 32x16 (KxN), row-major [K x ldb] source: lane l -> col = l&15, hi = l>>4;
// same K mapping as A (consistent with ISA A-16bit and sparse-B tables).
__device__ __forceinline__ v16h load_b_frag(const _Float16* B, int ldb, int k0, int n0) {
    int lane = threadIdx.x & 31;
    int n = lane & 15, hi = lane >> 4;
    v16h r;
#pragma unroll
    for (int i = 0; i < 8; ++i) {
        r[i]     = B[(size_t)(k0 + hi * 8 + i) * ldb + n0 + n];
        r[i + 8] = B[(size_t)(k0 + 16 + hi * 8 + i) * ldb + n0 + n];
    }
    return r;
}

#define WMMA_F16(a, b, c) \
    __builtin_amdgcn_wmma_f32_16x16x32_f16(false, (a), false, (b), (short)0, (c), false, false)

// ---------------- BatchNorm over (T,D) per node; one wave per node --------
__global__ void bn_kernel(const float* __restrict__ x, const float* __restrict__ g,
                          const float* __restrict__ b, _Float16* __restrict__ out) {
    int lane = threadIdx.x & 31;
    int node = blockIdx.x * (blockDim.x >> 5) + (threadIdx.x >> 5);
    if (node >= N_NODES) return;
    const float* xp = x + (size_t)node * TD;
    float s = 0.f, ss = 0.f;
    for (int i = lane; i < TD; i += 32) { float v = xp[i]; s += v; ss += v * v; }
#pragma unroll
    for (int o = 16; o > 0; o >>= 1) { s += __shfl_xor(s, o, 32); ss += __shfl_xor(ss, o, 32); }
    float mean = s * (1.f / TD);
    float var  = ss * (1.f / TD) - mean * mean;
    float rstd = rsqrtf(var + BN_EPS);
    float gg = g[node], bb = b[node];
    _Float16* op = out + (size_t)node * TD;
    for (int i = lane; i < TD; i += 32)
        op[i] = (_Float16)((xp[i] - mean) * rstd * gg + bb);
}

// ---------------- z = h16 @ W (64x64), WMMA; one wave per 16-row strip ----
__global__ void proj_kernel(const _Float16* __restrict__ h16, const float* __restrict__ W,
                            float* __restrict__ z) {
    __shared__ _Float16 Ws[D_FEAT * D_FEAT];
    for (int i = threadIdx.x; i < D_FEAT * D_FEAT; i += blockDim.x) Ws[i] = (_Float16)W[i];
    __syncthreads();
    int lane = threadIdx.x & 31;
    int n = lane & 15, hi = lane >> 4;
    int gw = (blockIdx.x * blockDim.x + threadIdx.x) >> 5;
    int nw = (gridDim.x * blockDim.x) >> 5;
    v16h bf[2][4];
#pragma unroll
    for (int kk = 0; kk < 2; ++kk)
#pragma unroll
        for (int j = 0; j < 4; ++j) bf[kk][j] = load_b_frag(Ws, D_FEAT, kk * 32, j * 16);
    const v8f zero8 = {0.f, 0.f, 0.f, 0.f, 0.f, 0.f, 0.f, 0.f};
    for (int s = gw; s < STRIPS; s += nw) {
        v8f acc[4];
#pragma unroll
        for (int j = 0; j < 4; ++j) acc[j] = zero8;
#pragma unroll
        for (int kk = 0; kk < 2; ++kk) {
            v16h a = load_a_frag(h16, D_FEAT, s * 16, kk * 32);
#pragma unroll
            for (int j = 0; j < 4; ++j) acc[j] = WMMA_F16(a, bf[kk][j], acc[j]);
        }
        size_t row0 = (size_t)s * 16;
#pragma unroll
        for (int j = 0; j < 4; ++j)
#pragma unroll
            for (int r = 0; r < 8; ++r)
                z[(row0 + r + hi * 8) * D_FEAT + j * 16 + n] = acc[j][r];
    }
}

// ---------------- el/er per (n,t,h) ---------------------------------------
__global__ void attn_coef_kernel(const float* __restrict__ z, const float* __restrict__ al,
                                 const float* __restrict__ ar, float* __restrict__ el,
                                 float* __restrict__ er) {
    int idx = blockIdx.x * blockDim.x + threadIdx.x;
    if (idx >= NTH) return;
    int h = idx & 3;
    const float* zp = z + (size_t)(idx >> 2) * D_FEAT + h * DH;
    float sl = 0.f, sr = 0.f;
#pragma unroll
    for (int e = 0; e < DH; ++e) { float v = zp[e]; sl += v * al[h * DH + e]; sr += v * ar[h * DH + e]; }
    el[idx] = sl;
    er[idx] = sr;
}

__global__ void init_kernel(unsigned* e1, unsigned* e2, float* d1, float* d2) {
    int idx = blockIdx.x * blockDim.x + threadIdx.x;
    if (idx >= NTH) return;
    e1[idx] = ORD_NEG_INF; e2[idx] = ORD_NEG_INF;
    d1[idx] = 0.f;         d2[idx] = 0.f;
}

// ---------------- edge pass 1: segment max --------------------------------
__global__ void edge_max_kernel(const int* __restrict__ src, const int* __restrict__ dst,
                                const float* __restrict__ el, const float* __restrict__ er,
                                unsigned* __restrict__ emax) {
    int idx = blockIdx.x * blockDim.x + threadIdx.x;
    if (idx >= ET) return;
    int e = idx / T_STEPS, t = idx - e * T_STEPS;
    int s = src[e], d = dst[e];
    const float* lp = el + ((size_t)s * T_STEPS + t) * H_HEADS;
    const float* rp = er + ((size_t)d * T_STEPS + t) * H_HEADS;
    unsigned* mp = emax + ((size_t)d * T_STEPS + t) * H_HEADS;
#pragma unroll
    for (int h = 0; h < H_HEADS; ++h) {
        float v = lp[h] + rp[h];
        v = v > 0.f ? v : SLOPE * v;
        atomicMax(&mp[h], f2ord(v));
    }
}

// ---------------- edge pass 2: segment sum of exp -------------------------
__global__ void edge_den_kernel(const int* __restrict__ src, const int* __restrict__ dst,
                                const float* __restrict__ el, const float* __restrict__ er,
                                const unsigned* __restrict__ emax, float* __restrict__ den) {
    int idx = blockIdx.x * blockDim.x + threadIdx.x;
    if (idx >= ET) return;
    int e = idx / T_STEPS, t = idx - e * T_STEPS;
    int s = src[e], d = dst[e];
    const float* lp = el + ((size_t)s * T_STEPS + t) * H_HEADS;
    const float* rp = er + ((size_t)d * T_STEPS + t) * H_HEADS;
    size_t dbase = ((size_t)d * T_STEPS + t) * H_HEADS;
#pragma unroll
    for (int h = 0; h < H_HEADS; ++h) {
        float v = lp[h] + rp[h];
        v = v > 0.f ? v : SLOPE * v;
        atomicAdd((float*)&den[dbase + h], expf(v - ord2f(emax[dbase + h])));
    }
}

// ---------------- edge pass 3: alpha * z[src] scatter-add into x2 ---------
__global__ void edge_agg_kernel(const int* __restrict__ src, const int* __restrict__ dst,
                                const float* __restrict__ el, const float* __restrict__ er,
                                const unsigned* __restrict__ emax, const float* __restrict__ den,
                                const float* __restrict__ z, float* __restrict__ x2) {
    int idx = blockIdx.x * blockDim.x + threadIdx.x;
    if (idx >= ET) return;
    int e = idx / T_STEPS, t = idx - e * T_STEPS;
    int s = src[e], d = dst[e];
    const float* lp = el + ((size_t)s * T_STEPS + t) * H_HEADS;
    const float* rp = er + ((size_t)d * T_STEPS + t) * H_HEADS;
    size_t dbase = ((size_t)d * T_STEPS + t) * H_HEADS;
    float alpha[H_HEADS];
#pragma unroll
    for (int h = 0; h < H_HEADS; ++h) {
        float v = lp[h] + rp[h];
        v = v > 0.f ? v : SLOPE * v;
        float ex = expf(v - ord2f(emax[dbase + h]));
        alpha[h] = ex / fmaxf(den[dbase + h], 1e-16f);
    }
    const float* zp = z + ((size_t)s * T_STEPS + t) * D_FEAT;
    float* xp = x2 + ((size_t)d * T_STEPS + t) * D_FEAT;
#pragma unroll
    for (int c = 0; c < D_FEAT; ++c)
        atomicAdd(&xp[c], alpha[c >> 4] * zp[c]);
}

// ---------------- fused FFN: out = x2 + gelu(h2@W1+b1)@W2 + b2 ------------
// blockDim must be 128 (4 waves); each wave owns a 16-row strip per iter.
// W1/W2 are pre-packed into fragment-order LDS (32B contiguous per lane per
// fragment) so each B fragment is two ds_load_b128 in the steady-state loop.
__global__ void ffn_kernel(const _Float16* __restrict__ h16, const float* __restrict__ w1,
                           const float* __restrict__ b1, const float* __restrict__ w2,
                           const float* __restrict__ b2, const float* __restrict__ x2,
                           float* __restrict__ out) {
    // frag f<16: W1 (kk=f>>3, j=f&7); frag f>=16: W2 (kk=(f-16)>>2, j=(f-16)&3)
    __shared__ __attribute__((aligned(32))) _Float16 B1p[16 * 32 * 16];  // 16 KB
    __shared__ __attribute__((aligned(32))) _Float16 B2p[16 * 32 * 16];  // 16 KB
    __shared__ __attribute__((aligned(32))) _Float16 Us[4][16 * DFF];    // 16 KB
    for (int u = threadIdx.x; u < 32 * 32; u += blockDim.x) {
        int f = u >> 5, l = u & 31;
        int pn = l & 15, phi = l >> 4;
        const float* srcW;
        _Float16* dstp;
        int ldb, k0, n0;
        if (f < 16) {
            srcW = w1; ldb = DFF;
            k0 = (f >> 3) * 32; n0 = (f & 7) * 16;
            dstp = &B1p[(size_t)u * 16];
        } else {
            int f2 = f - 16;
            srcW = w2; ldb = D_FEAT;
            k0 = (f2 >> 2) * 32; n0 = (f2 & 3) * 16;
            dstp = &B2p[((size_t)f2 * 32 + l) * 16];
        }
#pragma unroll
        for (int i = 0; i < 8; ++i) {
            dstp[i]     = (_Float16)srcW[(size_t)(k0 + phi * 8 + i) * ldb + n0 + pn];
            dstp[i + 8] = (_Float16)srcW[(size_t)(k0 + 16 + phi * 8 + i) * ldb + n0 + pn];
        }
    }
    __syncthreads();

    int lane = threadIdx.x & 31, wv = threadIdx.x >> 5;
    int n = lane & 15, hi = lane >> 4;
    int gw = (blockIdx.x * blockDim.x + threadIdx.x) >> 5;
    int nw = (gridDim.x * blockDim.x) >> 5;
    _Float16* U = &Us[wv][0];
    const v8f zero8 = {0.f, 0.f, 0.f, 0.f, 0.f, 0.f, 0.f, 0.f};
    for (int s = gw; s < STRIPS; s += nw) {
        // ---- u = h2(16x64) @ W1(64x128) ----
        v8f acc1[8];
#pragma unroll
        for (int j = 0; j < 8; ++j) acc1[j] = zero8;
#pragma unroll
        for (int kk = 0; kk < 2; ++kk) {
            v16h a = load_a_frag(h16, D_FEAT, s * 16, kk * 32);
#pragma unroll
            for (int j = 0; j < 8; ++j) {
                v16h bb = *(const v16h*)&B1p[(((size_t)(kk * 8 + j)) * 32 + lane) * 16];
                acc1[j] = WMMA_F16(a, bb, acc1[j]);
            }
        }
        // ---- bias + exact GELU, re-fragment through per-wave LDS tile ----
#pragma unroll
        for (int j = 0; j < 8; ++j) {
            float bias = b1[j * 16 + n];
#pragma unroll
            for (int r = 0; r < 8; ++r) {
                float u = acc1[j][r] + bias;
                float gl = 0.5f * u * (1.f + erff(u * 0.70710678118654752f));
                U[(r + hi * 8) * DFF + j * 16 + n] = (_Float16)gl;
            }
        }
        // per-wave private LDS region; DS ops are in-order within a wave.
        // ---- y = u(16x128) @ W2(128x64) ----
        v8f acc2[4];
#pragma unroll
        for (int j = 0; j < 4; ++j) acc2[j] = zero8;
#pragma unroll
        for (int kk = 0; kk < 4; ++kk) {
            v16h a2 = load_a_frag(U, DFF, 0, kk * 32);
#pragma unroll
            for (int j = 0; j < 4; ++j) {
                v16h bb = *(const v16h*)&B2p[(((size_t)(kk * 4 + j)) * 32 + lane) * 16];
                acc2[j] = WMMA_F16(a2, bb, acc2[j]);
            }
        }
        // ---- epilogue: + b2 + residual x2 ----
#pragma unroll
        for (int j = 0; j < 4; ++j) {
            float bias = b2[j * 16 + n];
#pragma unroll
            for (int r = 0; r < 8; ++r) {
                size_t row = (size_t)s * 16 + r + hi * 8;
                out[row * D_FEAT + j * 16 + n] =
                    x2[row * D_FEAT + j * 16 + n] + acc2[j][r] + bias;
            }
        }
    }
}

// --------------------------------------------------------------------------
extern "C" void kernel_launch(void* const* d_in, const int* in_sizes, int n_in,
                              void* d_out, int out_size, void* d_ws, size_t ws_size,
                              hipStream_t stream) {
    (void)in_sizes; (void)n_in; (void)out_size; (void)ws_size;
    const float* x    = (const float*)d_in[0];
    const int*   src1 = (const int*)d_in[1];
    const int*   dst1 = (const int*)d_in[2];
    const int*   src2 = (const int*)d_in[3];
    const int*   dst2 = (const int*)d_in[4];
    const float* W1   = (const float*)d_in[5];
    const float* al1  = (const float*)d_in[6];
    const float* ar1  = (const float*)d_in[7];
    const float* W2   = (const float*)d_in[8];
    const float* al2  = (const float*)d_in[9];
    const float* ar2  = (const float*)d_in[10];
    const float* bn1g = (const float*)d_in[11];
    const float* bn1b = (const float*)d_in[12];
    const float* bn2g = (const float*)d_in[13];
    const float* bn2b = (const float*)d_in[14];
    const float* fw1  = (const float*)d_in[15];
    const float* fb1  = (const float*)d_in[16];
    const float* fw2  = (const float*)d_in[17];
    const float* fb2  = (const float*)d_in[18];
    float* out = (float*)d_out;

    char* p = (char*)d_ws;
    auto carve = [&](size_t bytes) -> char* {
        char* r = p;
        p += (bytes + 255) & ~(size_t)255;
        return r;
    };
    _Float16* h16  = (_Float16*)carve(NTD * sizeof(_Float16));  // reused for h2
    float*    z1   = (float*)carve(NTD * sizeof(float));
    float*    z2   = (float*)carve(NTD * sizeof(float));
    float*    el1  = (float*)carve((size_t)NTH * 4);
    float*    er1  = (float*)carve((size_t)NTH * 4);
    float*    el2  = (float*)carve((size_t)NTH * 4);
    float*    er2  = (float*)carve((size_t)NTH * 4);
    unsigned* emax1 = (unsigned*)carve((size_t)NTH * 4);
    unsigned* emax2 = (unsigned*)carve((size_t)NTH * 4);
    float*    den1 = (float*)carve((size_t)NTH * 4);
    float*    den2 = (float*)carve((size_t)NTH * 4);
    float*    x2   = (float*)carve(NTD * sizeof(float));

    const int g_nth = (NTH + 255) / 256;
    const int g_et  = (ET + 255) / 256;

    // 1. BN1 -> h (f16)
    bn_kernel<<<N_NODES / 8, 256, 0, stream>>>(x, bn1g, bn1b, h16);
    // 2. projections (WMMA)
    proj_kernel<<<128, 256, 0, stream>>>(h16, W1, z1);
    proj_kernel<<<128, 256, 0, stream>>>(h16, W2, z2);
    // 3. attention coefficients
    attn_coef_kernel<<<g_nth, 256, 0, stream>>>(z1, al1, ar1, el1, er1);
    attn_coef_kernel<<<g_nth, 256, 0, stream>>>(z2, al2, ar2, el2, er2);
    // 4. init segment state + residual base x2 = x
    init_kernel<<<g_nth, 256, 0, stream>>>(emax1, emax2, den1, den2);
    hipMemcpyAsync(x2, x, NTD * sizeof(float), hipMemcpyDeviceToDevice, stream);
    // 5. segment softmax + aggregate (adds into x2 => fused residual)
    edge_max_kernel<<<g_et, 256, 0, stream>>>(src1, dst1, el1, er1, emax1);
    edge_max_kernel<<<g_et, 256, 0, stream>>>(src2, dst2, el2, er2, emax2);
    edge_den_kernel<<<g_et, 256, 0, stream>>>(src1, dst1, el1, er1, emax1, den1);
    edge_den_kernel<<<g_et, 256, 0, stream>>>(src2, dst2, el2, er2, emax2, den2);
    edge_agg_kernel<<<g_et, 256, 0, stream>>>(src1, dst1, el1, er1, emax1, den1, z1, x2);
    edge_agg_kernel<<<g_et, 256, 0, stream>>>(src2, dst2, el2, er2, emax2, den2, z2, x2);
    // 6. BN2 -> h2 (f16, reuse h16)
    bn_kernel<<<N_NODES / 8, 256, 0, stream>>>(x2, bn2g, bn2b, h16);
    // 7. fused FFN + residual -> out
    ffn_kernel<<<256, 128, 0, stream>>>(h16, fw1, fb1, fw2, fb2, x2, out);
}
